// Qwen3NextSparseMoeBlock_62019327754717
// MI455X (gfx1250) — matrix operand order repaired and checked by reading, not compiled, not run on hardware
//
#include <hip/hip_runtime.h>
#include <hip/hip_bf16.h>
#include <cstdint>

typedef __attribute__((ext_vector_type(16))) __bf16 v16bf;
typedef __attribute__((ext_vector_type(8)))  float  v8f;
typedef __attribute__((ext_vector_type(4)))  int    v4i;

// ---- fixed problem sizes ----
constexpr int Hc  = 2048;   // hidden
constexpr int Ec  = 8;      // experts
constexpr int Ic  = 1024;   // moe intermediate
constexpr int ISc = 4096;   // shared intermediate
constexpr int Tc  = 4096;   // B*S tokens

// ---- workspace layout (bytes, 256-aligned) ----
constexpr size_t OFF_COMBINE = 0;                                      // T*E f32
constexpr size_t OFF_GATE    = OFF_COMBINE + (size_t)Tc * Ec * 4;      // T   f32
constexpr size_t OFF_XB      = OFF_GATE    + (size_t)Tc * 4;           // T*H bf16
constexpr size_t OFF_GACT    = OFF_XB      + (size_t)Tc * Hc * 2;      // T*E*I bf16
constexpr size_t OFF_SG      = OFF_GACT    + (size_t)Tc * Ec * Ic * 2; // T*IS bf16

// ---- CDNA5 async global->LDS path (guarded: compiles either way) ----
#if defined(__has_builtin)
#if __has_builtin(__builtin_amdgcn_global_load_async_to_lds_b128)
#define ATHENA_ASYNC_LDS 1
#endif
#endif
#ifndef ATHENA_ASYNC_LDS
#define ATHENA_ASYNC_LDS 0
#endif

#if ATHENA_ASYNC_LDS
typedef __attribute__((address_space(1))) v4i* g_v4i_ptr;  // global 128b chunk
typedef __attribute__((address_space(3))) v4i* l_v4i_ptr;  // LDS 128b chunk
static __device__ inline void async_cp_b128(const void* g, void* l) {
  __builtin_amdgcn_global_load_async_to_lds_b128((g_v4i_ptr)(g), (l_v4i_ptr)(l),
                                                 0, 0);
}
static __device__ inline void wait_async0() {
#if __has_builtin(__builtin_amdgcn_s_wait_asynccnt)
  __builtin_amdgcn_s_wait_asynccnt(0);
#else
  asm volatile("s_wait_asynccnt 0x0" ::: "memory");
#endif
}
#endif

static __device__ inline float fsig(float x) {   // fast sigmoid: v_exp + v_rcp
  return __builtin_amdgcn_rcpf(1.f + __expf(-x));
}

// =====================================================================
// Router: logits, top-2 combine weights, shared-expert sigmoid gate dot
// =====================================================================
__global__ __launch_bounds__(256) void router_kernel(
    const float* __restrict__ x, const float* __restrict__ Wr,
    const float* __restrict__ Wsh, float* __restrict__ logits,
    float* __restrict__ combine, float* __restrict__ gate)
{
  __shared__ float lg[32][8];
  const int lt = threadIdx.x >> 3, e = threadIdx.x & 7;
  const int t = blockIdx.x * 32 + lt;
  const float* xr = x + (size_t)t * Hc;
  float acc = 0.f, ga = 0.f;
  for (int h = 0; h < Hc; ++h) {
    const float xv = xr[h];
    acc += xv * Wr[h * Ec + e];
    if (e == 0) ga += xv * Wsh[h];
  }
  lg[lt][e] = acc;
  logits[(size_t)t * Ec + e] = acc;
  if (e == 0) gate[t] = ga;
  __syncthreads();
  if (threadIdx.x < 32) {
    const int tt = blockIdx.x * 32 + threadIdx.x;
    float l[8];
#pragma unroll
    for (int i = 0; i < 8; ++i) l[i] = lg[threadIdx.x][i];
    int i0 = 0; float l0 = l[0];
#pragma unroll
    for (int i = 1; i < 8; ++i) if (l[i] > l0) { l0 = l[i]; i0 = i; }
    int i1 = -1; float l1 = -3.4e38f;
#pragma unroll
    for (int i = 0; i < 8; ++i) if (i != i0 && l[i] > l1) { l1 = l[i]; i1 = i; }
    // top-2 of softmax == top-2 of logits; renormalized pair weights:
    const float w0 = 1.f / (1.f + __expf(l1 - l0));
    const float w1 = 1.f - w0;
#pragma unroll
    for (int i = 0; i < 8; ++i)
      combine[(size_t)tt * Ec + i] = (i == i0) ? w0 : (i == i1) ? w1 : 0.f;
  }
}

// =====================================================================
// fp32 -> bf16 cast of the token activations (one pass, 4 elems/thread)
// =====================================================================
__global__ __launch_bounds__(256) void cvt_bf16_kernel(
    const float* __restrict__ in, __bf16* __restrict__ out)
{
  const size_t i = ((size_t)blockIdx.x * 256 + threadIdx.x) * 4;
  const float4 v = *(const float4*)(in + i);
  union { __bf16 h[4]; uint2 u; } p;
  p.h[0] = (__bf16)v.x; p.h[1] = (__bf16)v.y;
  p.h[2] = (__bf16)v.z; p.h[3] = (__bf16)v.w;
  *(uint2*)(out + i) = p.u;
}

// =====================================================================
// Tiled bf16 WMMA GEMM, block 128x128, 8 waves @ 32x64, K-step 32,
// software-pipelined: B prefetched into VGPRs, A copied via async-to-LDS.
//  MODE 0: MoE up    A=xb[T,H]    B=Wg_e/Wu_e[H,I]   -> gact bf16 = combine*silu(g)*u
//  MODE 1: MoE down  A=gact[T,EI] B=Wd[EI,H]         -> out f32 (write)
//  MODE 2: shared up A=xb[T,H]    B=Wsg/Wsu[H,IS]    -> sg bf16 = silu(g)*u
//  MODE 3: shared dn A=sg[T,IS]   B=Wsd[IS,H]        -> out f32 += v*sigmoid(gate)
// =====================================================================
union Frag { uint4 q[2]; v16bf v; };

static __device__ inline v8f wmma_bf16(v16bf a, v16bf b, v8f c) {
  return __builtin_amdgcn_wmma_f32_16x16x32_bf16(false, a, false, b, (short)0, c,
                                                 false, false);
}

template <int MODE>
__global__ __launch_bounds__(256) void moe_gemm(
    const __bf16* __restrict__ A, const float* __restrict__ B1g,
    const float* __restrict__ B2g, float* __restrict__ outF,
    __bf16* __restrict__ outB, const float* __restrict__ combine,
    const float* __restrict__ gate)
{
  constexpr bool DUAL = (MODE == 0 || MODE == 2);
  constexpr int NN = (MODE == 0) ? Ic : (MODE == 2) ? ISc : Hc;
  constexpr int KK = (MODE == 0 || MODE == 2) ? Hc : (MODE == 1) ? (Ec * Ic) : ISc;
  constexpr int BK = 32, LDP = 40;  // padded LDS row (bf16 elems)

  __shared__ __bf16 lA[128][LDP];                       // A tile, row-major [M][K]
  __shared__ __bf16 lB1[128][LDP];                      // B tile, transposed [N][K]
  __shared__ __bf16 lB2[DUAL ? 128 : 1][DUAL ? LDP : 1];

  const int tid = threadIdx.x;
  const int lane = tid & 31, wid = tid >> 5;
  const int waveM = wid >> 1, waveN = wid & 1;
  const int mBlk = blockIdx.y * 128, nBlk = blockIdx.x * 128;

  const float* B1 = B1g;
  const float* B2 = B2g;
  if constexpr (MODE == 0) {
    B1 += (size_t)blockIdx.z * Hc * Ic;
    B2 += (size_t)blockIdx.z * Hc * Ic;
  }

  const v8f zero = {0.f, 0.f, 0.f, 0.f, 0.f, 0.f, 0.f, 0.f};
  v8f acc1[2][4];
  v8f acc2[2][4];
#pragma unroll
  for (int i = 0; i < 2; ++i)
#pragma unroll
    for (int j = 0; j < 4; ++j) {
      acc1[i][j] = zero;
      if constexpr (DUAL) acc2[i][j] = zero;
    }

  // ---- staging indices ----
  const __bf16* aRow = A + (size_t)(mBlk + (tid >> 1)) * KK + (tid & 1) * 16;
  __bf16* lA0 = &lA[tid >> 1][(tid & 1) * 16];   // this thread's LDS A slot
  const int nLd = tid & 127, kqLd = (tid >> 7) & 1;

  const float* pB1 = B1 + (size_t)(kqLd * 16) * NN + nBlk + nLd;
  const float* pB2 = nullptr;
  if constexpr (DUAL) pB2 = B2 + (size_t)(kqLd * 16) * NN + nBlk + nLd;

  // ---- prologue prefetch (K-step 0) ----
  float rB1[16], rB2[16];
#pragma unroll
  for (int kk = 0; kk < 16; ++kk) rB1[kk] = pB1[(size_t)kk * NN];
  if constexpr (DUAL) {
#pragma unroll
    for (int kk = 0; kk < 16; ++kk) rB2[kk] = pB2[(size_t)kk * NN];
  }
#if !ATHENA_ASYNC_LDS
  uint4 rA0 = ((const uint4*)aRow)[0];
  uint4 rA1 = ((const uint4*)aRow)[1];
#endif

  for (int k0 = 0; k0 < KK; k0 += BK) {
    __syncthreads();  // prior reads done

    // ---- store prefetched B tile (f32 -> bf16, transposed [N][K]) ----
    {
      alignas(16) __bf16 t1[16];
#pragma unroll
      for (int kk = 0; kk < 16; ++kk) t1[kk] = (__bf16)rB1[kk];
      *(uint4*)&lB1[nLd][kqLd * 16]     = *(const uint4*)&t1[0];
      *(uint4*)&lB1[nLd][kqLd * 16 + 8] = *(const uint4*)&t1[8];
      if constexpr (DUAL) {
        alignas(16) __bf16 t2[16];
#pragma unroll
        for (int kk = 0; kk < 16; ++kk) t2[kk] = (__bf16)rB2[kk];
        *(uint4*)&lB2[nLd][kqLd * 16]     = *(const uint4*)&t2[0];
        *(uint4*)&lB2[nLd][kqLd * 16 + 8] = *(const uint4*)&t2[8];
      }
    }
    // ---- A tile: bf16 copy straight into LDS ----
#if ATHENA_ASYNC_LDS
    async_cp_b128(aRow + k0, lA0);
    async_cp_b128(aRow + k0 + 8, lA0 + 8);
#else
    *(uint4*)lA0       = rA0;
    *(uint4*)(lA0 + 8) = rA1;
#endif

    // ---- issue next K-step global loads (overlap with WMMA below) ----
    if (k0 + BK < KK) {
      pB1 += (size_t)BK * NN;
#pragma unroll
      for (int kk = 0; kk < 16; ++kk) rB1[kk] = pB1[(size_t)kk * NN];
      if constexpr (DUAL) {
        pB2 += (size_t)BK * NN;
#pragma unroll
        for (int kk = 0; kk < 16; ++kk) rB2[kk] = pB2[(size_t)kk * NN];
      }
#if !ATHENA_ASYNC_LDS
      rA0 = ((const uint4*)(aRow + k0 + BK))[0];
      rA1 = ((const uint4*)(aRow + k0 + BK))[1];
#endif
    }

#if ATHENA_ASYNC_LDS
    wait_async0();
#endif
    __syncthreads();  // tile visible

    // ---- fragments + 16 (dual) / 8 WMMAs ----
    Frag aF[2];
    const int aoff = (lane < 16) ? 0 : 16;
#pragma unroll
    for (int ms = 0; ms < 2; ++ms) {
      const int row = waveM * 32 + ms * 16 + (lane & 15);
      aF[ms].q[0] = *(const uint4*)&lA[row][aoff];
      aF[ms].q[1] = *(const uint4*)&lA[row][aoff + 8];
    }
    const int koffB = (lane < 16) ? 0 : 16;
#pragma unroll
    for (int ns = 0; ns < 4; ++ns) {
      const int col = waveN * 64 + ns * 16 + (lane & 15);
      Frag bF;
      bF.q[0] = *(const uint4*)&lB1[col][koffB];
      bF.q[1] = *(const uint4*)&lB1[col][koffB + 8];
#pragma unroll
      for (int ms = 0; ms < 2; ++ms)
        acc1[ms][ns] = wmma_bf16(aF[ms].v, bF.v, acc1[ms][ns]);
      if constexpr (DUAL) {
        Frag cF;
        cF.q[0] = *(const uint4*)&lB2[col][koffB];
        cF.q[1] = *(const uint4*)&lB2[col][koffB + 8];
#pragma unroll
        for (int ms = 0; ms < 2; ++ms)
          acc2[ms][ns] = wmma_bf16(aF[ms].v, cF.v, acc2[ms][ns]);
      }
    }
  }

  // epilogue: D layout -> VGPR r: M = r (lane<16) / 8+r (lane>=16), N = lane&15
  const int mW = mBlk + waveM * 32;
  const int nW = nBlk + waveN * 64;
  const int rAdd = (lane < 16) ? 0 : 8;
  const int colL = lane & 15;
#pragma unroll
  for (int ms = 0; ms < 2; ++ms) {
#pragma unroll
    for (int r = 0; r < 8; ++r) {
      const int row = mW + ms * 16 + rAdd + r;
      float sc = 1.f;
      if constexpr (MODE == 0) sc = combine[(size_t)row * Ec + blockIdx.z];
      if constexpr (MODE == 3) sc = fsig(gate[row]);
#pragma unroll
      for (int ns = 0; ns < 4; ++ns) {
        const int col = nW + ns * 16 + colL;
        const float g = acc1[ms][ns][r];
        if constexpr (MODE == 0) {
          const float u = acc2[ms][ns][r];
          const float v = g * fsig(g) * u * sc;
          outB[(size_t)row * (Ec * Ic) + (size_t)blockIdx.z * Ic + col] = (__bf16)v;
        } else if constexpr (MODE == 2) {
          const float u = acc2[ms][ns][r];
          outB[(size_t)row * ISc + col] = (__bf16)(g * fsig(g) * u);
        } else if constexpr (MODE == 1) {
          outF[(size_t)row * Hc + col] = g;
        } else {
          outF[(size_t)row * Hc + col] += g * sc;
        }
      }
    }
  }
}

// =====================================================================
extern "C" void kernel_launch(void* const* d_in, const int* in_sizes, int n_in,
                              void* d_out, int out_size, void* d_ws, size_t ws_size,
                              hipStream_t stream)
{
  (void)in_sizes; (void)n_in; (void)out_size; (void)ws_size;
  const float* x   = (const float*)d_in[0];
  const float* Wr  = (const float*)d_in[1];
  const float* Wg  = (const float*)d_in[2];
  const float* Wu  = (const float*)d_in[3];
  const float* Wd  = (const float*)d_in[4];   // [E,I,H] == stacked [E*I, H]
  const float* Wsg = (const float*)d_in[5];
  const float* Wsu = (const float*)d_in[6];
  const float* Wsd = (const float*)d_in[7];
  const float* Wsh = (const float*)d_in[8];

  float* out       = (float*)d_out;           // [T, H] final
  float* outLogits = out + (size_t)Tc * Hc;   // [T, E] router logits

  char*   ws      = (char*)d_ws;
  float*  combine = (float*)(ws + OFF_COMBINE);
  float*  gate    = (float*)(ws + OFF_GATE);
  __bf16* xb      = (__bf16*)(ws + OFF_XB);
  __bf16* gact    = (__bf16*)(ws + OFF_GACT);
  __bf16* sg      = (__bf16*)(ws + OFF_SG);

  router_kernel<<<Tc / 32, 256, 0, stream>>>(x, Wr, Wsh, outLogits, combine, gate);
  cvt_bf16_kernel<<<(int)(((size_t)Tc * Hc) / (256 * 4)), 256, 0, stream>>>(x, xb);

  // MoE up (per expert, fused gate+up, combine weight folded into activations)
  moe_gemm<0><<<dim3(Ic / 128, Tc / 128, Ec), 256, 0, stream>>>(
      xb, Wg, Wu, nullptr, gact, combine, nullptr);
  // MoE down: single K=E*I reduction GEMM -> writes final buffer
  moe_gemm<1><<<dim3(Hc / 128, Tc / 128, 1), 256, 0, stream>>>(
      gact, Wd, nullptr, out, nullptr, nullptr, nullptr);
  // Shared expert up
  moe_gemm<2><<<dim3(ISc / 128, Tc / 128, 1), 256, 0, stream>>>(
      xb, Wsg, Wsu, nullptr, sg, nullptr, nullptr);
  // Shared expert down, gated, accumulated into final buffer
  moe_gemm<3><<<dim3(Hc / 128, Tc / 128, 1), 256, 0, stream>>>(
      sg, Wsd, nullptr, out, nullptr, nullptr, gate);
}
